// Attn_45423574123081
// MI455X (gfx1250) — compile-verified, matrix operand order checked by reading, not run on hardware
//
#include <hip/hip_runtime.h>
#include <hip/hip_bf16.h>

typedef __attribute__((ext_vector_type(2)))  float  v2f;
typedef __attribute__((ext_vector_type(8)))  float  v8f;
typedef __attribute__((ext_vector_type(8)))  __bf16 v8bf;
typedef __attribute__((ext_vector_type(16))) __bf16 v16bf;

#define B_SZ 32
#define L_SZ 1024
#define D_SZ 64

// Build a 16-bit A/B fragment (16x32 slice, one 32-wide K chunk) from a
// row-major bf16 row of 64 K values.
// ISA layout: lane<16 -> K = c*32 + {0..7} in v0..3, {16..23} in v4..7;
//             lane>=16 -> K = c*32 + {8..15} and {24..31}.
__device__ inline v16bf load_frag_bf16(const __bf16* row, int half, int chunk) {
    const __bf16* p = row + chunk * 32 + 8 * half;
    v8bf lo8 = *(const v8bf*)(p);
    v8bf hi8 = *(const v8bf*)(p + 16);
    v16bf f;
#pragma unroll
    for (int i = 0; i < 8; ++i) { f[i] = lo8[i]; f[i + 8] = hi8[i]; }
    return f;
}

// Split a row-major fp32 row (64 K values) into bf16 hi/lo fragments.
__device__ inline void split_frag_f32(const float* row, int half, int chunk,
                                      v16bf& fh, v16bf& fl) {
    const float* p = row + chunk * 32 + 8 * half;
#pragma unroll
    for (int i = 0; i < 16; ++i) {
        float x = p[(i < 8) ? i : (i + 8)];
        __bf16 h = (__bf16)x;
        fh[i] = h;
        fl[i] = (__bf16)(x - (float)h);
    }
}

// ---------------------------------------------------------------------------
// Pass 1: energy[b,j,e] = sum_d enc[b,j,d] * W[e,d] + bias[e], computed with
// exact f32 WMMA, stored as bf16 hi/lo split (~16 effective mantissa bits).
// One wave per (batch, 16-row j tile). K-step outer loop with 4 concurrent
// e-tile accumulators so B loads batch into clauses instead of serializing.
// ---------------------------------------------------------------------------
__global__ void energy_kernel(const float* __restrict__ enc,
                              const float* __restrict__ W,
                              const float* __restrict__ bias,
                              __bf16* __restrict__ ehi,
                              __bf16* __restrict__ elo) {
    int b    = blockIdx.x >> 6;        // L/16 = 64 tiles per batch
    int jt   = blockIdx.x & 63;
    int j0   = jt * 16;
    int lane = threadIdx.x;
    int lm   = lane & 15;
    int half = lane >> 4;              // f32 A: K = {4s,4s+1} | {4s+2,4s+3}

    const float* arow = enc + ((size_t)b * L_SZ + j0 + lm) * D_SZ + 2 * half;
    v2f a[16];
#pragma unroll
    for (int s = 0; s < 16; ++s) a[s] = *(const v2f*)(arow + 4 * s);

    const float* brow0 = W + (size_t)lm * D_SZ + 2 * half;  // + et*16*D per tile
    v8f acc[4] = {v8f{}, v8f{}, v8f{}, v8f{}};
#pragma unroll
    for (int s = 0; s < 16; ++s) {
        v2f bf[4];
#pragma unroll
        for (int et = 0; et < 4; ++et)
            bf[et] = *(const v2f*)(brow0 + (size_t)et * 16 * D_SZ + 4 * s);
#pragma unroll
        for (int et = 0; et < 4; ++et)
            acc[et] = __builtin_amdgcn_wmma_f32_16x16x4_f32(
                false, a[s], false, bf[et], (short)0, acc[et], false, false);
    }

#pragma unroll
    for (int et = 0; et < 4; ++et) {
        int e0 = et * 16;
        float bb = bias[e0 + lm];
#pragma unroll
        for (int r = 0; r < 8; ++r) {
            float v = acc[et][r] + bb;
            size_t idx = ((size_t)b * L_SZ + j0 + r + 8 * half) * D_SZ + e0 + lm;
            __bf16 h = (__bf16)v;
            ehi[idx] = h;
            elo[idx] = (__bf16)(v - (float)h);
        }
    }
}

// ---------------------------------------------------------------------------
// Pass 2: scores[t,b,j] = sum_d energy[b,j,d] * enc[b,t,d] via split-bf16
// WMMA (ahi*bhi + ahi*blo + alo*bhi, f32 accumulate); mask diag; softmax
// over j; write out[t,b,j]. Block = 8 waves per (16-row t tile, batch);
// wave w covers j in [w*128, w*128+128).
// ---------------------------------------------------------------------------
__global__ void scores_softmax_kernel(const float* __restrict__ enc,
                                      const __bf16* __restrict__ ehi,
                                      const __bf16* __restrict__ elo,
                                      float* __restrict__ out) {
    int tt   = blockIdx.x;             // t tile
    int b    = blockIdx.y;
    int t0   = tt * 16;
    int tid  = threadIdx.x;
    int w    = tid >> 5;
    int lane = tid & 31;
    int lm   = lane & 15;
    int half = lane >> 4;

    __shared__ float redmax[8][16];
    __shared__ float redsum[8][16];

    // A fragments from enc rows t0..t0+15, split to bf16 hi/lo once per wave
    const float* arow = enc + ((size_t)b * L_SZ + t0 + lm) * D_SZ;
    v16bf ah[2], al[2];
#pragma unroll
    for (int c = 0; c < 2; ++c) split_frag_f32(arow, half, c, ah[c], al[c]);

    // GEMM: 8 j-tiles x 2 k-chunks x 3 v_wmma_f32_16x16x32_bf16
    v8f cacc[8];
#pragma unroll
    for (int jt = 0; jt < 8; ++jt) {
        int j0 = w * 128 + jt * 16;
        size_t rowoff = ((size_t)b * L_SZ + j0 + lm) * D_SZ;
        const __bf16* bhrow = ehi + rowoff;
        const __bf16* blrow = elo + rowoff;
        v8f acc = {};
#pragma unroll
        for (int c = 0; c < 2; ++c) {
            v16bf bh = load_frag_bf16(bhrow, half, c);
            v16bf bl = load_frag_bf16(blrow, half, c);
            acc = __builtin_amdgcn_wmma_f32_16x16x32_bf16(
                false, al[c], false, bh, (short)0, acc, false, false);
            acc = __builtin_amdgcn_wmma_f32_16x16x32_bf16(
                false, ah[c], false, bl, (short)0, acc, false, false);
            acc = __builtin_amdgcn_wmma_f32_16x16x32_bf16(
                false, ah[c], false, bh, (short)0, acc, false, false);
        }
        cacc[jt] = acc;
    }

    // Diagonal mask: scores[t,:,t] = 0  (C layout: M = r + 8*half, N = lm)
#pragma unroll
    for (int jt = 0; jt < 8; ++jt) {
        int j_g = w * 128 + jt * 16 + lm;
#pragma unroll
        for (int r = 0; r < 8; ++r) {
            int t_g = t0 + r + 8 * half;
            if (j_g == t_g) cacc[jt][r] = 0.0f;
        }
    }

    // Row max: per-lane, xor-shuffle across the 16-lane group, cross-wave LDS
    float rmax[8];
#pragma unroll
    for (int r = 0; r < 8; ++r) {
        float m = cacc[0][r];
#pragma unroll
        for (int jt = 1; jt < 8; ++jt) m = fmaxf(m, cacc[jt][r]);
#pragma unroll
        for (int msk = 1; msk <= 8; msk <<= 1)
            m = fmaxf(m, __shfl_xor(m, msk, 32));
        rmax[r] = m;
    }
    if (lm == 0) {
#pragma unroll
        for (int r = 0; r < 8; ++r) redmax[w][r + 8 * half] = rmax[r];
    }
    __syncthreads();
    float ginv[8];  // global max now, inverse sum later
#pragma unroll
    for (int r = 0; r < 8; ++r) {
        float m = redmax[0][r + 8 * half];
#pragma unroll
        for (int ww = 1; ww < 8; ++ww) m = fmaxf(m, redmax[ww][r + 8 * half]);
        ginv[r] = m;
    }

    // exp(x - max) in-register + row sums
    float rsum[8];
#pragma unroll
    for (int r = 0; r < 8; ++r) rsum[r] = 0.0f;
#pragma unroll
    for (int jt = 0; jt < 8; ++jt) {
#pragma unroll
        for (int r = 0; r < 8; ++r) {
            float e = __expf(cacc[jt][r] - ginv[r]);
            cacc[jt][r] = e;
            rsum[r] += e;
        }
    }
#pragma unroll
    for (int r = 0; r < 8; ++r) {
        float s = rsum[r];
#pragma unroll
        for (int msk = 1; msk <= 8; msk <<= 1)
            s += __shfl_xor(s, msk, 32);
        rsum[r] = s;
    }
    if (lm == 0) {
#pragma unroll
        for (int r = 0; r < 8; ++r) redsum[w][r + 8 * half] = rsum[r];
    }
    __syncthreads();
#pragma unroll
    for (int r = 0; r < 8; ++r) {
        float s = redsum[0][r + 8 * half];
#pragma unroll
        for (int ww = 1; ww < 8; ++ww) s += redsum[ww][r + 8 * half];
        ginv[r] = 1.0f / s;
    }

    // Write out[t, b, j] — lanes 0..15 / 16..31 store coalesced 64B runs
#pragma unroll
    for (int r = 0; r < 8; ++r) {
        size_t obase = ((size_t)(t0 + r + 8 * half) * B_SZ + b) * L_SZ;
#pragma unroll
        for (int jt = 0; jt < 8; ++jt) {
            int j_g = w * 128 + jt * 16 + lm;
            out[obase + j_g] = cacc[jt][r] * ginv[r];
        }
    }
}

extern "C" void kernel_launch(void* const* d_in, const int* in_sizes, int n_in,
                              void* d_out, int out_size, void* d_ws, size_t ws_size,
                              hipStream_t stream) {
    const float* enc  = (const float*)d_in[0];   // (B, L, D) fp32
    const float* W    = (const float*)d_in[1];   // (D, D) fp32
    const float* bias = (const float*)d_in[2];   // (D,) fp32
    // d_in[3] = target_length (device scalar); T derived from out_size.
    float* out = (float*)d_out;

    __bf16* ehi = (__bf16*)d_ws;                     // B*L*D bf16 = 4 MB
    __bf16* elo = ehi + (size_t)B_SZ * L_SZ * D_SZ;  // + 4 MB

    int T = out_size / (B_SZ * L_SZ);                // = 1024 here

    energy_kernel<<<dim3(B_SZ * (L_SZ / 16)), dim3(32), 0, stream>>>(
        enc, W, bias, ehi, elo);
    scores_softmax_kernel<<<dim3(T / 16, B_SZ), dim3(256), 0, stream>>>(
        enc, ehi, elo, out);
}
